// Generator_66494683677172
// MI455X (gfx1250) — compile-verified, matrix operand order checked by reading, not run on hardware
//
#include <hip/hip_runtime.h>
#include <stdint.h>

// Problem constants (from the reference)
constexpr int Bn = 16, T = 256, V = 18000, NS = 30, L = 10, G = 3, H = 400;
constexpr int N  = NS * Bn;   // 480 rows (slot-major: n = slot*B + batch)
constexpr int H3 = 3 * H;     // 1200
constexpr int Hp = 416;       // H padded to a multiple of 32 for WMMA K

typedef __attribute__((ext_vector_type(16))) __bf16 v16bf;
typedef __attribute__((ext_vector_type(8)))  float v8f;

static __device__ __forceinline__ unsigned short f2bf(float f) {
    uint32_t u = __float_as_uint(f);
    u += 0x7FFFu + ((u >> 16) & 1u);   // round-to-nearest-even
    return (unsigned short)(u >> 16);
}
static __device__ __forceinline__ float sigmoidf_(float x) {
    return 1.0f / (1.0f + expf(-x));
}

// f32 (rows x H) -> bf16 (rows x Hp), zero-padding columns H..Hp-1
__global__ void k_convert_pad(const float* __restrict__ src,
                              unsigned short* __restrict__ dst, int rows) {
    int i = blockIdx.x * blockDim.x + threadIdx.x;
    if (i >= rows * Hp) return;
    int r = i / Hp, c = i - r * Hp;
    dst[i] = (c < H) ? f2bf(src[(size_t)r * H + c]) : (unsigned short)0;
}

// combined[slot] = slot_emb[domain_idx[slot]] + slot_emb[slotname_idx[slot]]
__global__ void k_combined(const float* __restrict__ se, const int* __restrict__ dom,
                           const int* __restrict__ slo, float* __restrict__ comb) {
    int i = blockIdx.x * blockDim.x + threadIdx.x;
    if (i >= NS * H) return;
    int s = i / H, k = i - s * H;
    comb[i] = se[(size_t)dom[s] * H + k] + se[(size_t)slo[s] * H + k];
}

// h0[n] = encoded_hidden[0][n%B]; x0[n] = combined[n/B]
__global__ void k_init(const float* __restrict__ eh, const float* __restrict__ comb,
                       float* __restrict__ h, float* __restrict__ x) {
    int i = blockIdx.x * blockDim.x + threadIdx.x;
    if (i >= N * H) return;
    int n = i / H, k = i - n * H;
    h[i] = eh[(size_t)(n & 15) * H + k];
    x[i] = comb[(size_t)(n >> 4) * H + k];
}

// teacher forcing: x[n] = emb[target_batches[n%B, n/B, sPrev]]
__global__ void k_embed(const float* __restrict__ emb, const int* __restrict__ tb,
                        float* __restrict__ x, int sPrev) {
    int i = blockIdx.x * blockDim.x + threadIdx.x;
    if (i >= N * H) return;
    int n = i / H, k = i - n * H;
    int b = n & 15, sl = n >> 4;
    int tok = tb[((size_t)b * NS + sl) * L + sPrev];
    x[i] = emb[(size_t)tok * H + k];
}

// C[M x Nc] = A[M x K] * Bt[Nc x K]^T   (bf16 row-major, K multiple of 32, K <= Hp)
// One block (4 waves) = one 16-row M tile x 16 N tiles.
// A tile staged in LDS via global_load_async_to_lds_b128 (ASYNCcnt path).
// Each wave accumulates 4 independent 16x16 tiles, branch-free WMMA inner loop
// (tail tiles are clamped for loads and only the stores are guarded).
__global__ void k_gemm(const unsigned short* __restrict__ A,
                       const unsigned short* __restrict__ Bt,
                       float* __restrict__ C, int M, int Nc, int K, int ldc) {
    __shared__ __align__(16) unsigned short As[16 * Hp];   // 13.3 KB

    int tilesN = Nc >> 4;
    int nbm = (tilesN + 15) >> 4;            // blocks per M tile row
    int tm = blockIdx.x / nbm;
    int tnBase = (blockIdx.x - tm * nbm) << 4;

    // --- async stage A tile (16 x K bf16) into LDS -------------------------
    {
        const int bytes = 16 * K * 2;
        const char* ag = (const char*)(A + (size_t)tm * 16 * K);
        for (int o = threadIdx.x * 16; o < bytes; o += blockDim.x * 16) {
            uint32_t laddr = (uint32_t)(uintptr_t)((const char*)As + o);
            uint64_t gaddr = (uint64_t)(uintptr_t)(ag + o);
            asm volatile("global_load_async_to_lds_b128 %0, %1, off"
                         :: "v"(laddr), "v"(gaddr) : "memory");
        }
        asm volatile("s_wait_asynccnt 0x0" ::: "memory");
    }
    __syncthreads();

    int lane = threadIdx.x & 31;
    int wv   = __builtin_amdgcn_readfirstlane(threadIdx.x >> 5);  // scalar wave id
    int row  = lane & 15;
    int kb   = (lane >> 4) << 3;             // 0 or 8 (ISA 16-bit A/B layout)

    int tn0 = tn0 = tnBase + wv * 4;
    const uint32_t* brow[4];
    #pragma unroll
    for (int j = 0; j < 4; ++j) {
        int tn = tn0 + j;
        if (tn > tilesN - 1) tn = tilesN - 1;          // clamp tail (in-bounds)
        brow[j] = (const uint32_t*)(Bt + (size_t)(tn * 16 + row) * K);
    }
    const uint32_t* arow = (const uint32_t*)(As + row * K);  // LDS

    v8f acc[4] = {};
    for (int k0 = 0; k0 < K; k0 += 32) {
        int base = (k0 + kb) >> 1;           // dword index of bf16 pair
        union Frag { uint32_t u[8]; v16bf v; };
        Frag fa;
        #pragma unroll
        for (int i = 0; i < 4; ++i) {
            fa.u[i]     = arow[base + i];    // K = kb + 2i, 2i+1
            fa.u[4 + i] = arow[base + 8 + i];
        }
        #pragma unroll
        for (int j = 0; j < 4; ++j) {
            Frag fb;
            #pragma unroll
            for (int i = 0; i < 4; ++i) {
                fb.u[i]     = brow[j][base + i];
                fb.u[4 + i] = brow[j][base + 8 + i];
            }
            acc[j] = __builtin_amdgcn_wmma_f32_16x16x32_bf16(
                false, fa.v, false, fb.v, (short)0, acc[j], false, false);
        }
    }

    int mb = tm * 16 + ((lane >> 4) << 3);   // C layout: vgpr r -> M = mb + r
    #pragma unroll
    for (int j = 0; j < 4; ++j) {
        if (tn0 + j < tilesN) {              // uniform (scalar) guard
            int cn = (tn0 + j) * 16 + row;
            #pragma unroll
            for (int r = 0; r < 8; ++r)
                C[(size_t)(mb + r) * ldc + cn] = acc[j][r];
        }
    }
}

// Fused GRU cell (elementwise), h updated in place
__global__ void k_gru(const float* __restrict__ gi, const float* __restrict__ gh,
                      const float* __restrict__ bih, const float* __restrict__ bhh,
                      float* __restrict__ h) {
    int i = blockIdx.x * blockDim.x + threadIdx.x;
    if (i >= N * H) return;
    int n = i / H, k = i - n * H;
    size_t o = (size_t)n * H3;
    float ir = gi[o + k]         + bih[k],         hr = gh[o + k]         + bhh[k];
    float iz = gi[o + H + k]     + bih[H + k],     hz = gh[o + H + k]     + bhh[H + k];
    float in_ = gi[o + 2*H + k]  + bih[2*H + k],   hn = gh[o + 2*H + k]   + bhh[2*H + k];
    float r = sigmoidf_(ir + hr);
    float z = sigmoidf_(iz + hz);
    float nn = tanhf(in_ + r * hn);
    h[i] = (1.0f - z) * nn + z * h[i];
}

// scores[n,t] = dot(h[n], enc[n%B, t]) with length mask; one wave per (n,t)
__global__ void k_scores(const float* __restrict__ h, const float* __restrict__ enc,
                         const int* __restrict__ lens, float* __restrict__ sc) {
    int w = blockIdx.x * (blockDim.x >> 5) + (threadIdx.x >> 5);
    if (w >= N * T) return;
    int n = w / T, t = w - n * T;
    int lane = threadIdx.x & 31;
    const float* hv = h + (size_t)n * H;
    const float* ev = enc + ((size_t)(n & 15) * T + t) * H;
    float s = 0.f;
    for (int k = lane; k < H; k += 32) s += hv[k] * ev[k];
    #pragma unroll
    for (int o = 16; o; o >>= 1) s += __shfl_xor(s, o, 32);
    if (lane == 0) sc[w] = (t < lens[n & 15]) ? s : -1e9f;
}

// in-place softmax over T (one block of T threads per row)
__global__ void k_softmax_T(float* __restrict__ sc) {
    __shared__ float red[T];
    int n = blockIdx.x, t = threadIdx.x;
    float v = sc[(size_t)n * T + t];
    red[t] = v; __syncthreads();
    for (int o = 128; o; o >>= 1) { if (t < o) red[t] = fmaxf(red[t], red[t + o]); __syncthreads(); }
    float m = red[0]; __syncthreads();
    float e = expf(v - m);
    red[t] = e; __syncthreads();
    for (int o = 128; o; o >>= 1) { if (t < o) red[t] += red[t + o]; __syncthreads(); }
    sc[(size_t)n * T + t] = e / red[0];
}

// context[n,k] = sum_t prob[n,t] * enc[n%B, t, k]
__global__ void k_context(const float* __restrict__ prob, const float* __restrict__ enc,
                          float* __restrict__ ctx) {
    int i = blockIdx.x * blockDim.x + threadIdx.x;
    if (i >= N * H) return;
    int n = i / H, k = i - n * H;
    const float* pr = prob + (size_t)n * T;
    const float* eb = enc + (size_t)(n & 15) * T * H + k;
    float c = 0.f;
    for (int t = 0; t < T; ++t) c += pr[t] * eb[(size_t)t * H];
    ctx[i] = c;
}

// switch[n] = sigmoid([h, ctx, x] . w_ratio + b_ratio); one wave per n
__global__ void k_switch(const float* __restrict__ h, const float* __restrict__ ctx,
                         const float* __restrict__ x, const float* __restrict__ wr,
                         const float* __restrict__ br, float* __restrict__ sw) {
    int n = blockIdx.x * (blockDim.x >> 5) + (threadIdx.x >> 5);
    if (n >= N) return;
    int lane = threadIdx.x & 31;
    size_t o = (size_t)n * H;
    float s = 0.f;
    for (int k = lane; k < H; k += 32)
        s += h[o + k] * wr[k] + ctx[o + k] * wr[H + k] + x[o + k] * wr[2 * H + k];
    #pragma unroll
    for (int of = 16; of; of >>= 1) s += __shfl_xor(s, of, 32);
    if (lane == 0) sw[n] = sigmoidf_(s + br[0]);
}

// in-place row softmax over V, then scale by switch[n] (final = switch * p_vocab)
__global__ void k_softmax_v(float* __restrict__ base, const float* __restrict__ sw, int ldc) {
    __shared__ float red[256];
    int n = blockIdx.x, t = threadIdx.x;
    float* row = base + (size_t)n * ldc;
    float m = -3.4e38f;
    for (int v = t; v < V; v += 256) m = fmaxf(m, row[v]);
    red[t] = m; __syncthreads();
    for (int o = 128; o; o >>= 1) { if (t < o) red[t] = fmaxf(red[t], red[t + o]); __syncthreads(); }
    m = red[0]; __syncthreads();
    float ssum = 0.f;
    for (int v = t; v < V; v += 256) ssum += expf(row[v] - m);
    red[t] = ssum; __syncthreads();
    for (int o = 128; o; o >>= 1) { if (t < o) red[t] += red[t + o]; __syncthreads(); }
    float scale = sw[n] / red[0];
    for (int v = t; v < V; v += 256) row[v] = expf(row[v] - m) * scale;
}

// pointer distribution scatter: out[n, story[n%B, t]] += (1-switch[n]) * prob[n,t]
__global__ void k_scatter(const float* __restrict__ prob, const int* __restrict__ story,
                          const float* __restrict__ sw, float* __restrict__ base, int ldc) {
    int i = blockIdx.x * blockDim.x + threadIdx.x;
    if (i >= N * T) return;
    int n = i / T, t = i - n * T;
    int v = story[(size_t)(n & 15) * T + t];
    atomicAdd(base + (size_t)n * ldc + v, (1.0f - sw[n]) * prob[i]);
}

// gates[n,g] = context0[n] . w_gate[g] + b_gate[g]
__global__ void k_gates(const float* __restrict__ ctx, const float* __restrict__ wg,
                        const float* __restrict__ bg, float* __restrict__ outg) {
    int i = blockIdx.x * blockDim.x + threadIdx.x;
    if (i >= N * G) return;
    int n = i / G, g = i - n * G;
    const float* cv = ctx + (size_t)n * H;
    const float* wv = wg + (size_t)g * H;
    float a = bg[g];
    for (int k = 0; k < H; ++k) a += cv[k] * wv[k];
    outg[(size_t)n * G + g] = a;
}

extern "C" void kernel_launch(void* const* d_in, const int* in_sizes, int n_in,
                              void* d_out, int out_size, void* d_ws, size_t ws_size,
                              hipStream_t stream) {
    const float* eh   = (const float*)d_in[0];
    const float* enc  = (const float*)d_in[1];
    const float* emb  = (const float*)d_in[2];
    const float* wih  = (const float*)d_in[3];
    const float* whh  = (const float*)d_in[4];
    const float* bih  = (const float*)d_in[5];
    const float* bhh  = (const float*)d_in[6];
    const float* wr   = (const float*)d_in[7];
    const float* br   = (const float*)d_in[8];
    const float* wg   = (const float*)d_in[9];
    const float* bg   = (const float*)d_in[10];
    const float* se   = (const float*)d_in[11];
    const int* lens   = (const int*)d_in[12];
    const int* story  = (const int*)d_in[13];
    const int* tb     = (const int*)d_in[14];
    const int* dom    = (const int*)d_in[15];
    const int* slo    = (const int*)d_in[16];
    float* out = (float*)d_out;

    char* ws = (char*)d_ws;
    size_t off = 0;
    auto alloc = [&](size_t bytes) -> void* {
        void* p = ws + off;
        off += (bytes + 255) & ~(size_t)255;
        return p;
    };
    unsigned short* emb_bf = (unsigned short*)alloc((size_t)V * Hp * 2);
    unsigned short* wih_bf = (unsigned short*)alloc((size_t)H3 * Hp * 2);
    unsigned short* whh_bf = (unsigned short*)alloc((size_t)H3 * Hp * 2);
    float* comb = (float*)alloc((size_t)NS * H * 4);
    float* xf   = (float*)alloc((size_t)N * H * 4);
    float* hf   = (float*)alloc((size_t)N * H * 4);
    unsigned short* xbf = (unsigned short*)alloc((size_t)N * Hp * 2);
    unsigned short* hbf = (unsigned short*)alloc((size_t)N * Hp * 2);
    float* gi   = (float*)alloc((size_t)N * H3 * 4);
    float* gh   = (float*)alloc((size_t)N * H3 * 4);
    float* prob = (float*)alloc((size_t)N * T * 4);
    float* ctx  = (float*)alloc((size_t)N * H * 4);
    float* sw   = (float*)alloc((size_t)N * 4);

    const int TPB = 256;
    auto cdiv = [](long long a, long long b) { return (int)((a + b - 1) / b); };

    // one-time conversions / setup
    k_convert_pad<<<cdiv((long long)V  * Hp, TPB), TPB, 0, stream>>>(emb, emb_bf, V);
    k_convert_pad<<<cdiv((long long)H3 * Hp, TPB), TPB, 0, stream>>>(wih, wih_bf, H3);
    k_convert_pad<<<cdiv((long long)H3 * Hp, TPB), TPB, 0, stream>>>(whh, whh_bf, H3);
    k_combined<<<cdiv(NS * H, TPB), TPB, 0, stream>>>(se, dom, slo, comb);
    k_init<<<cdiv(N * H, TPB), TPB, 0, stream>>>(eh, comb, hf, xf);

    // grid: (M/16) * ceil((Nc/16)/16) blocks of 128 threads
    const int grid_g = (N / 16) * cdiv(H3 / 16, 16);   // 30 * 5
    const int grid_v = (N / 16) * cdiv(V  / 16, 16);   // 30 * 71

    for (int s = 0; s < L; ++s) {
        if (s > 0)
            k_embed<<<cdiv(N * H, TPB), TPB, 0, stream>>>(emb, tb, xf, s - 1);
        k_convert_pad<<<cdiv(N * Hp, TPB), TPB, 0, stream>>>(xf, xbf, N);
        k_convert_pad<<<cdiv(N * Hp, TPB), TPB, 0, stream>>>(hf, hbf, N);
        // GRU gate GEMMs (WMMA bf16)
        k_gemm<<<grid_g, 128, 0, stream>>>(xbf, wih_bf, gi, N, H3, Hp, H3);
        k_gemm<<<grid_g, 128, 0, stream>>>(hbf, whh_bf, gh, N, H3, Hp, H3);
        k_gru<<<cdiv(N * H, TPB), TPB, 0, stream>>>(gi, gh, bih, bhh, hf);
        k_convert_pad<<<cdiv(N * Hp, TPB), TPB, 0, stream>>>(hf, hbf, N);
        // attention
        k_scores<<<cdiv(N * T, 8), 256, 0, stream>>>(hf, enc, lens, prob);
        k_softmax_T<<<N, T, 0, stream>>>(prob);
        k_context<<<cdiv(N * H, TPB), TPB, 0, stream>>>(prob, enc, ctx);
        if (s == 0)
            k_gates<<<cdiv(N * G, TPB), TPB, 0, stream>>>(ctx, wg, bg, out + (size_t)N * L * V);
        k_switch<<<cdiv(N, 8), 256, 0, stream>>>(hf, ctx, xf, wr, br, sw);
        // vocab projection straight into the output slab (row n at (n*L+s)*V)
        float* obase = out + (size_t)s * V;
        k_gemm<<<grid_v, 128, 0, stream>>>(hbf, emb_bf, obase, N, V, Hp, L * V);
        k_softmax_v<<<N, 256, 0, stream>>>(obase, sw, L * V);
        k_scatter<<<cdiv(N * T, TPB), TPB, 0, stream>>>(prob, story, sw, obase, L * V);
    }
}
